// DecoderOnlyWrapper_47734266528430
// MI455X (gfx1250) — compile-verified, matrix operand order checked by reading, not run on hardware
//
#include <hip/hip_runtime.h>

typedef unsigned short u16;
typedef unsigned int   u32;

typedef __attribute__((ext_vector_type(16))) __bf16 v16bf_t;
typedef __attribute__((ext_vector_type(8)))  float  v8f_t;

union Frag { v16bf_t f; uint4 q[2]; };
union Acc  { v8f_t v; float f[8]; };

#define Bn 4
#define Qn 1024
#define Dn 2048
#define Hn 16
#define KVn 8
#define HDn 128
#define MAXn 2048
#define DFFn 5632
#define Ln 2
#define PASTn 1024
#define Mn (Bn*Qn) // 4096 tokens

__device__ __forceinline__ u16 f2bf(float f) {
    u32 u = __float_as_uint(f);
    u32 r = u + 0x7FFFu + ((u >> 16) & 1u);   // round-to-nearest-even
    return (u16)(r >> 16);
}

// ---------------------------------------------------------------------------
// Tiled WMMA GEMM: C[M,N] = A[M,K](bf16) * Bw[K,N](f32 -> bf16 on load)
// EPI: 0 = store f32, 1 = accumulate into f32 C, 2 = bf16( silu(acc) * Uaux )
// Block 256 thr (8 waves), tile 128x128, BK=64, double-buffered global
// prefetch into registers. Wave grid 2(M)x4(N), 4x2 accum of 16x16 per wave,
// 16 WMMAs per wave per LDS stage.
// ---------------------------------------------------------------------------
template<int EPI>
__global__ __launch_bounds__(256)
void gemm_bf16_kernel(const u16* __restrict__ A, const float* __restrict__ Bw,
                      float* __restrict__ C, const float* __restrict__ Uaux,
                      u16* __restrict__ Obf, int N, int K)
{
    __shared__ u32 sA[128][36];  // [m][k/2] rows 144B (16B aligned), 64 bf16 + pad
    __shared__ u16 sB[128][72];  // [n][k]   rows 144B (16B aligned), transposed

    const int tid  = threadIdx.x;
    const int lane = tid & 31;
    const int wid  = tid >> 5;
    const int lo   = lane & 15;
    const int hi   = lane >> 4;
    const int bm   = blockIdx.y * 128;
    const int bn   = blockIdx.x * 128;
    const int wm   = (wid >> 2) * 64;
    const int wn   = (wid & 3) * 32;

    const int arow = tid >> 1, aseg = tid & 1;   // A loader: 32 bf16 / thread
    const int bkk  = tid >> 2, bnb = tid & 3;    // B loader: 32 f32  / thread

    const u16*   aptr = A  + (size_t)(bm + arow) * K + aseg * 32;
    const float* bptr = Bw + (size_t)bkk * N + bn + bnb * 32;

    // prefetch tile k0 = 0 into registers
    uint4  aP[4];
    float4 bP[8];
    {
        const uint4* ag = (const uint4*)aptr;
#pragma unroll
        for (int i = 0; i < 4; ++i) aP[i] = ag[i];
        const float4* bg = (const float4*)bptr;
#pragma unroll
        for (int i = 0; i < 8; ++i) bP[i] = bg[i];
    }

    Acc acc[4][2];
#pragma unroll
    for (int i = 0; i < 4; ++i)
#pragma unroll
        for (int j = 0; j < 2; ++j)
#pragma unroll
            for (int r = 0; r < 8; ++r) acc[i][j].f[r] = 0.f;

    for (int k0 = 0; k0 < K; k0 += 64) {
        // commit staged registers to LDS
#pragma unroll
        for (int i = 0; i < 4; ++i)
            *(uint4*)&sA[arow][aseg * 16 + i * 4] = aP[i];
#pragma unroll
        for (int q = 0; q < 8; ++q) {
            float4 v = bP[q];
            int nb = bnb * 32 + q * 4;
            sB[nb + 0][bkk] = f2bf(v.x);
            sB[nb + 1][bkk] = f2bf(v.y);
            sB[nb + 2][bkk] = f2bf(v.z);
            sB[nb + 3][bkk] = f2bf(v.w);
        }
        __syncthreads();

        // issue global prefetch of next tile before compute (latency hiding)
        if (k0 + 64 < K) {
            const uint4* ag = (const uint4*)(aptr + k0 + 64);
#pragma unroll
            for (int i = 0; i < 4; ++i) aP[i] = ag[i];
            const float4* bg = (const float4*)(bptr + (size_t)(k0 + 64) * N);
#pragma unroll
            for (int i = 0; i < 8; ++i) bP[i] = bg[i];
        }

#pragma unroll
        for (int t = 0; t < 2; ++t) {           // two 32-deep k-steps per stage
            Frag af[4], bf[2];
#pragma unroll
            for (int i = 0; i < 4; ++i) {
                const u32* p = &sA[wm + i * 16 + lo][0];
                af[i].q[0] = *(const uint4*)(p + t * 16 + hi * 4);
                af[i].q[1] = *(const uint4*)(p + t * 16 + 8 + hi * 4);
            }
#pragma unroll
            for (int j = 0; j < 2; ++j) {
                const u16* p = &sB[wn + j * 16 + lo][0];
                bf[j].q[0] = *(const uint4*)(p + t * 32 + hi * 8);
                bf[j].q[1] = *(const uint4*)(p + t * 32 + 16 + hi * 8);
            }
#pragma unroll
            for (int i = 0; i < 4; ++i)
#pragma unroll
                for (int j = 0; j < 2; ++j)
                    acc[i][j].v = __builtin_amdgcn_wmma_f32_16x16x32_bf16(
                        false, af[i].f, false, bf[j].f, (short)0, acc[i][j].v, false, false);
        }
        __syncthreads();
    }

#pragma unroll
    for (int i = 0; i < 4; ++i)
#pragma unroll
        for (int j = 0; j < 2; ++j)
#pragma unroll
            for (int r = 0; r < 8; ++r) {
                int row = bm + wm + i * 16 + r + hi * 8;
                int col = bn + wn + j * 16 + lo;
                size_t g = (size_t)row * N + col;
                float v = acc[i][j].f[r];
                if (EPI == 0) {
                    C[g] = v;
                } else if (EPI == 1) {
                    C[g] += v;
                } else {
                    float u = Uaux[g];
                    float s = v / (1.f + __expf(-v));       // silu
                    Obf[g] = f2bf(s * u);
                }
            }
}

// ---------------------------------------------------------------------------
// Flash attention (WMMA, bf16 inputs, f32 online softmax).
// grid (Q/64, B*H), block 128 (4 waves). Wave = 16 q rows. 128-key tiles are
// staged in LDS, computed as two 64-key half-tiles to keep live VGPRs < 256
// (sc[4] instead of sc[8]) so fragment double-buffering survives regalloc.
// ---------------------------------------------------------------------------
__global__ __launch_bounds__(128)
void flash_attn_kernel(const u16* __restrict__ Qb,   // [B][H][Q][HD]
                       const u16* __restrict__ Kb,   // [B][KV][MAX][HD]
                       const u16* __restrict__ Vb,   // [B][KV][MAX][HD]
                       u16* __restrict__ Obf)        // [B][Q][H*HD]
{
    __shared__ u16 sK [128][136];   // [s][d]   rows 272B (16B aligned)
    __shared__ u16 sVT[128][136];   // [d][s]   transposed
    __shared__ u16 sP [4][16][72];  // per-wave P staging [m][64 keys + pad]

    const int tid  = threadIdx.x;
    const int lane = tid & 31;
    const int w    = tid >> 5;
    const int lo   = lane & 15;
    const int hi   = lane >> 4;
    const int qt   = blockIdx.x;
    const int bh   = blockIdx.y;
    const int b    = bh / Hn, h = bh % Hn, kv = h / (Hn / KVn);
    const int q0   = qt * 64 + w * 16;
    const float scale = 0.08838834764831845f; // 1/sqrt(128)

    // Q fragments (held in registers for whole kernel)
    Frag qf[4];
    {
        const u16* qrow = Qb + ((size_t)(b * Hn + h) * Qn + q0 + lo) * HDn;
#pragma unroll
        for (int kk = 0; kk < 4; ++kk) {
            qf[kk].q[0] = *(const uint4*)(qrow + kk * 32 + hi * 8);
            qf[kk].q[1] = *(const uint4*)(qrow + kk * 32 + 16 + hi * 8);
        }
    }

    float mrun[8], lrun[8];
    Acc oacc[8];
#pragma unroll
    for (int r = 0; r < 8; ++r) { mrun[r] = -1e30f; lrun[r] = 0.f; }
#pragma unroll
    for (int d = 0; d < 8; ++d)
#pragma unroll
        for (int r = 0; r < 8; ++r) oacc[d].f[r] = 0.f;

    const int smax   = PASTn + qt * 64 + 64;     // keys needed by this q-block
    const int ntiles = (smax + 127) >> 7;
    const u16* Kbase = Kb + (size_t)(b * KVn + kv) * MAXn * HDn;
    const u16* Vbase = Vb + (size_t)(b * KVn + kv) * MAXn * HDn;

    for (int st = 0; st < ntiles; ++st) {
        // cooperative stage: thread tid loads key/value row s = st*128+tid
        {
            const uint4* kg = (const uint4*)(Kbase + (size_t)(st * 128 + tid) * HDn);
            const uint4* vg = (const uint4*)(Vbase + (size_t)(st * 128 + tid) * HDn);
#pragma unroll
            for (int q8 = 0; q8 < 16; ++q8) {
                uint4 kx = kg[q8];
                *(uint4*)&sK[tid][q8 * 8] = kx;
                uint4 vx = vg[q8];
                u16* vs = (u16*)&vx;
#pragma unroll
                for (int e = 0; e < 8; ++e) sVT[q8 * 8 + e][tid] = vs[e];
            }
        }
        __syncthreads();

#pragma unroll
        for (int half = 0; half < 2; ++half) {   // 64-key compute half-tiles
            const int sbase = st * 128 + half * 64;
            if (sbase >= smax) break;            // causal: nothing left

            // scores: 16 x 64; K-fragments double-buffered
            Acc sc[4];
#pragma unroll
            for (int j = 0; j < 4; ++j)
#pragma unroll
                for (int r = 0; r < 8; ++r) sc[j].f[r] = 0.f;
#pragma unroll
            for (int kk = 0; kk < 4; ++kk) {
                Frag kfb[2];
                {
                    const u16* p = &sK[half * 64 + lo][0];
                    kfb[0].q[0] = *(const uint4*)(p + kk * 32 + hi * 8);
                    kfb[0].q[1] = *(const uint4*)(p + kk * 32 + 16 + hi * 8);
                }
#pragma unroll
                for (int j = 0; j < 4; ++j) {
                    int cur = j & 1;
                    if (j < 3) {
                        const u16* p = &sK[half * 64 + (j + 1) * 16 + lo][0];
                        kfb[cur ^ 1].q[0] = *(const uint4*)(p + kk * 32 + hi * 8);
                        kfb[cur ^ 1].q[1] = *(const uint4*)(p + kk * 32 + 16 + hi * 8);
                    }
                    sc[j].v = __builtin_amdgcn_wmma_f32_16x16x32_bf16(
                        false, qf[kk].f, false, kfb[cur].f, (short)0, sc[j].v, false, false);
                }
            }

            // causal mask + online softmax (row stats via 16-lane xor reduce)
            float alpha[8];
#pragma unroll
            for (int r = 0; r < 8; ++r) {
                int qpos = PASTn + q0 + r + hi * 8;
                float tmax = -1e30f;
#pragma unroll
                for (int j = 0; j < 4; ++j) {
                    int spos = sbase + j * 16 + lo;
                    float v = sc[j].f[r] * scale;
                    if (spos > qpos) v = -1e30f;
                    sc[j].f[r] = v;
                    tmax = fmaxf(tmax, v);
                }
#pragma unroll
                for (int off = 1; off < 16; off <<= 1)
                    tmax = fmaxf(tmax, __shfl_xor(tmax, off, 32));
                float mnew = fmaxf(mrun[r], tmax);
                alpha[r] = __expf(mrun[r] - mnew);
                float ls = 0.f;
#pragma unroll
                for (int j = 0; j < 4; ++j) {
                    float p = __expf(sc[j].f[r] - mnew);
                    sc[j].f[r] = p;
                    ls += p;
                }
#pragma unroll
                for (int off = 1; off < 16; off <<= 1)
                    ls += __shfl_xor(ls, off, 32);
                lrun[r] = lrun[r] * alpha[r] + ls;
                mrun[r] = mnew;
            }
#pragma unroll
            for (int d = 0; d < 8; ++d)
#pragma unroll
                for (int r = 0; r < 8; ++r) oacc[d].f[r] *= alpha[r];

            // relayout P (C-layout -> A-layout) through per-wave LDS staging
#pragma unroll
            for (int j = 0; j < 4; ++j)
#pragma unroll
                for (int r = 0; r < 8; ++r)
                    sP[w][r + hi * 8][j * 16 + lo] = f2bf(sc[j].f[r]);
            asm volatile("s_wait_dscnt 0" ::: "memory");   // same-wave LDS RAW

            // O += P @ V  (2 k-steps of 32 keys; V-fragments double-buffered)
#pragma unroll
            for (int kk2 = 0; kk2 < 2; ++kk2) {
                Frag pf;
                const u16* pp = &sP[w][lo][0];
                pf.q[0] = *(const uint4*)(pp + kk2 * 32 + hi * 8);
                pf.q[1] = *(const uint4*)(pp + kk2 * 32 + 16 + hi * 8);
                Frag vfb[2];
                {
                    const u16* vp = &sVT[lo][0];
                    vfb[0].q[0] = *(const uint4*)(vp + half * 64 + kk2 * 32 + hi * 8);
                    vfb[0].q[1] = *(const uint4*)(vp + half * 64 + kk2 * 32 + 16 + hi * 8);
                }
#pragma unroll
                for (int d = 0; d < 8; ++d) {
                    int cur = d & 1;
                    if (d < 7) {
                        const u16* vp = &sVT[(d + 1) * 16 + lo][0];
                        vfb[cur ^ 1].q[0] = *(const uint4*)(vp + half * 64 + kk2 * 32 + hi * 8);
                        vfb[cur ^ 1].q[1] = *(const uint4*)(vp + half * 64 + kk2 * 32 + 16 + hi * 8);
                    }
                    oacc[d].v = __builtin_amdgcn_wmma_f32_16x16x32_bf16(
                        false, pf.f, false, vfb[cur].f, (short)0, oacc[d].v, false, false);
                }
            }
        }
        __syncthreads();
    }

    // normalize and store to [B][Q][H*HD] (bf16, feeds Wo GEMM)
#pragma unroll
    for (int d = 0; d < 8; ++d)
#pragma unroll
        for (int r = 0; r < 8; ++r) {
            int qrow = q0 + r + hi * 8;
            float o = oacc[d].f[r] / lrun[r];
            Obf[(size_t)(b * Qn + qrow) * (Hn * HDn) + h * HDn + d * 16 + lo] = f2bf(o);
        }
}

// ---------------------------------------------------------------------------
// Elementwise / small kernels
// ---------------------------------------------------------------------------
__global__ void embed_kernel(const int* __restrict__ ids, const float* __restrict__ E,
                             float* __restrict__ Hout)
{
    int token = blockIdx.x;
    int id = ids[token];
    const float4* src = (const float4*)(E + (size_t)id * Dn);
    float4* dst = (float4*)(Hout + (size_t)token * Dn);
    for (int i = threadIdx.x; i < Dn / 4; i += blockDim.x) dst[i] = src[i];
}

__global__ __launch_bounds__(256)
void rmsnorm_kernel(const float* __restrict__ X, const float* __restrict__ W,
                    u16* __restrict__ Ybf, float* __restrict__ Yf32)
{
    __shared__ float red[256];
    int row = blockIdx.x;
    const float* x = X + (size_t)row * Dn;
    float ss = 0.f;
    for (int i = threadIdx.x; i < Dn; i += 256) { float v = x[i]; ss += v * v; }
    red[threadIdx.x] = ss;
    __syncthreads();
    for (int s = 128; s > 0; s >>= 1) {
        if (threadIdx.x < s) red[threadIdx.x] += red[threadIdx.x + s];
        __syncthreads();
    }
    float inv = rsqrtf(red[0] / (float)Dn + 1e-6f);
    for (int i = threadIdx.x; i < Dn; i += 256) {
        float y = x[i] * W[i] * inv;
        if (Ybf) Ybf[(size_t)row * Dn + i] = f2bf(y);
        else     Yf32[(size_t)row * Dn + i] = y;
    }
}

__global__ void rope_q_kernel(const float* __restrict__ Qf, u16* __restrict__ Qbf)
{
    int token = blockIdx.x;
    int b = token >> 10, qi = token & 1023;
    float pos = (float)(PASTn + qi);
    for (int idx = threadIdx.x; idx < Hn * 64; idx += blockDim.x) {
        int hh = idx >> 6, d = idx & 63;
        float inv = __expf(-(float)d * (9.210340371976184f / 64.f)); // 10000^(-d/64)
        float s, c;
        __sincosf(pos * inv, &s, &c);
        const float* src = Qf + (size_t)token * (Hn * HDn) + hh * HDn;
        float x1 = src[d], x2 = src[d + 64];
        u16* dst = Qbf + ((size_t)(b * Hn + hh) * Qn + qi) * HDn;
        dst[d]      = f2bf(x1 * c - x2 * s);
        dst[d + 64] = f2bf(x2 * c + x1 * s);
    }
}

__global__ void rope_k_kernel(const float* __restrict__ Kf, float* __restrict__ OutK)
{
    int token = blockIdx.x;
    int b = token >> 10, qi = token & 1023;
    float pos = (float)(PASTn + qi);
    for (int idx = threadIdx.x; idx < KVn * 64; idx += blockDim.x) {
        int kvh = idx >> 6, d = idx & 63;
        float inv = __expf(-(float)d * (9.210340371976184f / 64.f));
        float s, c;
        __sincosf(pos * inv, &s, &c);
        const float* src = Kf + (size_t)token * (KVn * HDn) + kvh * HDn;
        float x1 = src[d], x2 = src[d + 64];
        float* dst = OutK + ((size_t)(b * KVn + kvh) * MAXn + PASTn + qi) * HDn;
        dst[d]      = x1 * c - x2 * s;
        dst[d + 64] = x2 * c + x1 * s;
    }
}

__global__ void copy_v_kernel(const float* __restrict__ Vf, float* __restrict__ OutV)
{
    int token = blockIdx.x;
    int b = token >> 10, qi = token & 1023;
    for (int idx = threadIdx.x; idx < KVn * HDn; idx += blockDim.x) {
        int kvh = idx >> 7, d = idx & 127;
        OutV[((size_t)(b * KVn + kvh) * MAXn + PASTn + qi) * HDn + d] =
            Vf[(size_t)token * (KVn * HDn) + idx];
    }
}

__global__ void conv_bf16_kernel(const float* __restrict__ X, u16* __restrict__ Y)
{
    size_t i = (size_t)blockIdx.x * blockDim.x + threadIdx.x;
    Y[i] = f2bf(X[i]);
}

// ---------------------------------------------------------------------------
// Orchestration
// ---------------------------------------------------------------------------
extern "C" void kernel_launch(void* const* d_in, const int* in_sizes, int n_in,
                              void* d_out, int out_size, void* d_ws, size_t ws_size,
                              hipStream_t stream)
{
    (void)in_sizes; (void)n_in; (void)out_size; (void)ws_size;

    const int*   input_ids = (const int*)  d_in[0];
    const float* cache_k   = (const float*)d_in[3];
    const float* cache_v   = (const float*)d_in[4];
    const float* embed     = (const float*)d_in[5];
    const float* Wq        = (const float*)d_in[6];
    const float* Wk        = (const float*)d_in[7];
    const float* Wv        = (const float*)d_in[8];
    const float* Wo        = (const float*)d_in[9];
    const float* ln1       = (const float*)d_in[10];
    const float* ln2       = (const float*)d_in[11];
    const float* Wg        = (const float*)d_in[12];
    const float* Wu        = (const float*)d_in[13];
    const float* Wd        = (const float*)d_in[14];
    const float* norm_w    = (const float*)d_in[15];

    float* h    = (float*)d_out;                             // [B,Q,D]
    float* outk = h + (size_t)Mn * Dn;                       // [L,B,KV,MAX,HD]
    float* outv = outk + (size_t)Ln * Bn * KVn * MAXn * HDn; // [L,B,KV,MAX,HD]

    char* ws = (char*)d_ws;
    u16*   hn   = (u16*)ws;   ws += (size_t)Mn * Dn * 2;            // rmsnorm out (bf16)
    float* qf   = (float*)ws; ws += (size_t)Mn * Hn * HDn * 4;      // pre-RoPE Q
    float* kf   = (float*)ws; ws += (size_t)Mn * KVn * HDn * 4;     // pre-RoPE K
    float* vf   = (float*)ws; ws += (size_t)Mn * KVn * HDn * 4;     // V
    u16*   qbf  = (u16*)ws;   ws += (size_t)Mn * Hn * HDn * 2;      // RoPE'd Q (bf16)
    u16*   kbf  = (u16*)ws;   ws += (size_t)Bn * KVn * MAXn * HDn * 2; // full K cache bf16
    u16*   vbf  = (u16*)ws;   ws += (size_t)Bn * KVn * MAXn * HDn * 2; // full V cache bf16
    u16*   obf  = (u16*)ws;   ws += (size_t)Mn * Hn * HDn * 2;      // attn out (bf16)
    float* uf   = (float*)ws; ws += (size_t)Mn * DFFn * 4;          // up-proj f32
    u16*   actb = (u16*)ws;   ws += (size_t)Mn * DFFn * 2;          // silu(g)*u bf16

    const size_t perL = (size_t)Bn * KVn * MAXn * HDn;
    hipMemcpyAsync(outk, cache_k, (size_t)Ln * perL * sizeof(float),
                   hipMemcpyDeviceToDevice, stream);
    hipMemcpyAsync(outv, cache_v, (size_t)Ln * perL * sizeof(float),
                   hipMemcpyDeviceToDevice, stream);

    embed_kernel<<<Mn, 256, 0, stream>>>(input_ids, embed, h);

    for (int l = 0; l < Ln; ++l) {
        rmsnorm_kernel<<<Mn, 256, 0, stream>>>(h, ln1 + (size_t)l * Dn, hn, nullptr);

        gemm_bf16_kernel<0><<<dim3(Hn * HDn / 128, Mn / 128), 256, 0, stream>>>(
            hn, Wq + (size_t)l * Dn * Hn * HDn, qf, nullptr, nullptr, Hn * HDn, Dn);
        gemm_bf16_kernel<0><<<dim3(KVn * HDn / 128, Mn / 128), 256, 0, stream>>>(
            hn, Wk + (size_t)l * Dn * KVn * HDn, kf, nullptr, nullptr, KVn * HDn, Dn);
        gemm_bf16_kernel<0><<<dim3(KVn * HDn / 128, Mn / 128), 256, 0, stream>>>(
            hn, Wv + (size_t)l * Dn * KVn * HDn, vf, nullptr, nullptr, KVn * HDn, Dn);

        rope_q_kernel<<<Mn, 256, 0, stream>>>(qf, qbf);
        rope_k_kernel<<<Mn, 256, 0, stream>>>(kf, outk + (size_t)l * perL);
        copy_v_kernel<<<Mn, 256, 0, stream>>>(vf, outv + (size_t)l * perL);
        conv_bf16_kernel<<<(unsigned)(perL / 256), 256, 0, stream>>>(outk + (size_t)l * perL, kbf);
        conv_bf16_kernel<<<(unsigned)(perL / 256), 256, 0, stream>>>(outv + (size_t)l * perL, vbf);

        flash_attn_kernel<<<dim3(Qn / 64, Bn * Hn), 128, 0, stream>>>(qbf, kbf, vbf, obf);

        gemm_bf16_kernel<1><<<dim3(Dn / 128, Mn / 128), 256, 0, stream>>>(
            obf, Wo + (size_t)l * Hn * HDn * Dn, h, nullptr, nullptr, Dn, Hn * HDn);

        rmsnorm_kernel<<<Mn, 256, 0, stream>>>(h, ln2 + (size_t)l * Dn, hn, nullptr);

        gemm_bf16_kernel<0><<<dim3(DFFn / 128, Mn / 128), 256, 0, stream>>>(
            hn, Wu + (size_t)l * Dn * DFFn, uf, nullptr, nullptr, DFFn, Dn);
        gemm_bf16_kernel<2><<<dim3(DFFn / 128, Mn / 128), 256, 0, stream>>>(
            hn, Wg + (size_t)l * Dn * DFFn, nullptr, uf, actb, DFFn, Dn);
        gemm_bf16_kernel<1><<<dim3(Dn / 128, Mn / 128), 256, 0, stream>>>(
            actb, Wd + (size_t)l * DFFn * Dn, h, nullptr, nullptr, Dn, DFFn);
    }

    rmsnorm_kernel<<<Mn, 256, 0, stream>>>(h, norm_w, nullptr, h);
}
